// RegionMHA_28570122453883
// MI455X (gfx1250) — matrix operand order repaired and checked by reading, not compiled, not run on hardware
//
#include <hip/hip_runtime.h>

// ---------------------------------------------------------------------------
// RegionMHA for MI455X (gfx1250): bf16 WMMA GEMMs (async-LDS double-buffered)
// + flash attention. S=1024, N=8, D=1024, H=16, DK=64, SCALE=8.
// ---------------------------------------------------------------------------

typedef __attribute__((ext_vector_type(16))) __bf16 v16bf;
typedef __attribute__((ext_vector_type(8)))  float  v8f;

union BFrag {
    v16bf          v;
    unsigned short u[16];
    uint4          q[2];
};

__device__ __forceinline__ unsigned short f2bf(float f) {
    unsigned int x = __builtin_bit_cast(unsigned int, f);
    x += 0x7FFFu + ((x >> 16) & 1u);          // round-to-nearest-even
    return (unsigned short)(x >> 16);
}

__device__ __forceinline__ v8f wmma_bf16(const BFrag& a, const BFrag& b, v8f c) {
    return __builtin_amdgcn_wmma_f32_16x16x32_bf16(false, a.v, false, b.v,
                                                   (short)0, c, false, false);
}

// Async copy of 32 contiguous bytes global->LDS per lane (2x b128, ASYNCcnt+=2
// per wave). INST_OFFSET applies to both LDS and global addresses (ISA 10.x).
__device__ __forceinline__ void async_cp32(unsigned lds, const unsigned short* g) {
    unsigned long long ga = (unsigned long long)g;
    asm volatile("global_load_async_to_lds_b128 %0, %1, off\n\t"
                 "global_load_async_to_lds_b128 %0, %1, off offset:16"
                 :: "v"(lds), "v"(ga) : "memory");
}

#define DMODEL 1024
#define LDA    40          // LDS row stride (elements); 80B rows keep 16B align

// ---------------------------------------------------------------------------
// f32 -> bf16 conversion pass (one-time; removes cvt VALU from GEMM hot loops)
// ---------------------------------------------------------------------------
__global__ __launch_bounds__(256) void cvt_f32_bf16(const float* __restrict__ src,
                                                    unsigned short* __restrict__ dst,
                                                    int n4) {
    int i = blockIdx.x * blockDim.x + threadIdx.x;
    const int stride = gridDim.x * blockDim.x;
    for (; i < n4; i += stride) {
        float4 f = ((const float4*)src)[i];
        uint2 o;
        o.x = (unsigned)f2bf(f.x) | ((unsigned)f2bf(f.y) << 16);
        o.y = (unsigned)f2bf(f.z) | ((unsigned)f2bf(f.w) << 16);
        ((uint2*)dst)[i] = o;
    }
}

// ---------------------------------------------------------------------------
// GEMM core: Y[m,n] = sum_k A[m,k]*B[n,k] (+bias). A,B bf16, k-contiguous.
// Block tile 128x128, 256 threads (8 waves 2x4), wave tile 64x32 = 4x2 WMMA.
// LDS double-buffered; tiles staged with global_load_async_to_lds_b128 and
// synchronized via s_wait_asynccnt (copy of tile i+1 overlaps WMMA of tile i).
// ---------------------------------------------------------------------------
#define GEMM_PROLOG()                                                          \
    __shared__ unsigned short lA[2][128 * LDA];                                \
    __shared__ unsigned short lB[2][128 * LDA];                                \
    const int tid  = threadIdx.x;                                              \
    const int bm   = blockIdx.y * 128;                                         \
    const int bn   = blockIdx.x * 128;                                         \
    const int wave = tid >> 5;                                                 \
    const int lane = tid & 31;                                                 \
    const int half = lane >> 4;                                                \
    const int l16  = lane & 15;                                                \
    const int wm   = (wave & 1) * 64;                                          \
    const int wn   = (wave >> 1) * 32;                                         \
    const int srow = tid >> 1;                                                 \
    const int skb  = (tid & 1) * 16;                                           \
    const unsigned short* gA = A + (size_t)(bm + srow) * DMODEL + skb;         \
    const unsigned short* gB = B + (size_t)(bn + srow) * DMODEL + skb;         \
    unsigned aAddr[2], bAddr[2];                                               \
    aAddr[0] = (unsigned)(size_t)&lA[0][srow * LDA + skb];                     \
    aAddr[1] = (unsigned)(size_t)&lA[1][srow * LDA + skb];                     \
    bAddr[0] = (unsigned)(size_t)&lB[0][srow * LDA + skb];                     \
    bAddr[1] = (unsigned)(size_t)&lB[1][srow * LDA + skb];                     \
    v8f acc[4][2] = {};                                                        \
    async_cp32(aAddr[0], gA);                                                  \
    async_cp32(bAddr[0], gB);                                                  \
    int p = 0;                                                                 \
    for (int k0 = 0; k0 < DMODEL; k0 += 32) {                                  \
        const int kn = (k0 + 32) & (DMODEL - 1);   /* wrap: uniform counter */ \
        async_cp32(aAddr[1 - p], gA + kn);                                     \
        async_cp32(bAddr[1 - p], gB + kn);                                     \
        asm volatile("s_wait_asynccnt 4" ::: "memory"); /* tile k0 complete */ \
        __syncthreads();                                                       \
        const unsigned short* bufA = lA[p];                                    \
        const unsigned short* bufB = lB[p];                                    \
        BFrag af[4], bfr[2];                                                   \
        _Pragma("unroll")                                                      \
        for (int mi = 0; mi < 4; ++mi) {                                       \
            const unsigned short* q = &bufA[(wm + mi * 16 + l16) * LDA];       \
            af[mi].q[0] = *(const uint4*)(q + half * 16);                      \
            af[mi].q[1] = *(const uint4*)(q + half * 16 + 8);                  \
        }                                                                      \
        _Pragma("unroll")                                                      \
        for (int ni = 0; ni < 2; ++ni) {                                       \
            const unsigned short* q =                                          \
                &bufB[(wn + ni * 16 + l16) * LDA + half * 16];                 \
            bfr[ni].q[0] = *(const uint4*)(q);                                 \
            bfr[ni].q[1] = *(const uint4*)(q + 8);                             \
        }                                                                      \
        _Pragma("unroll")                                                      \
        for (int mi = 0; mi < 4; ++mi)                                         \
            _Pragma("unroll")                                                  \
            for (int ni = 0; ni < 2; ++ni)                                     \
                acc[mi][ni] = wmma_bf16(af[mi], bfr[ni], acc[mi][ni]);         \
        __syncthreads();                                                       \
        p ^= 1;                                                                \
    }

// Projection GEMM: bf16 output, plain row-major [8192,1024].
__global__ __launch_bounds__(256) void gemm_proj(const unsigned short* __restrict__ A,
                                                 const unsigned short* __restrict__ B,
                                                 const float* __restrict__ bias,
                                                 unsigned short* __restrict__ Out) {
    GEMM_PROLOG()
#pragma unroll
    for (int mi = 0; mi < 4; ++mi)
#pragma unroll
        for (int ni = 0; ni < 2; ++ni) {
            const int col = bn + wn + ni * 16 + l16;
            const float bv = bias[col];
#pragma unroll
            for (int v = 0; v < 8; ++v) {
                const int row = bm + wm + mi * 16 + half * 8 + v;
                Out[(size_t)row * DMODEL + col] = f2bf(acc[mi][ni][v] + bv);
            }
        }
}

// Output GEMM: rows m = nb*S + s, f32 output transposed to [S,N,D].
__global__ __launch_bounds__(256) void gemm_fc(const unsigned short* __restrict__ A,
                                               const unsigned short* __restrict__ B,
                                               const float* __restrict__ bias,
                                               float* __restrict__ Out) {
    GEMM_PROLOG()
#pragma unroll
    for (int mi = 0; mi < 4; ++mi)
#pragma unroll
        for (int ni = 0; ni < 2; ++ni) {
            const int col = bn + wn + ni * 16 + l16;
            const float bv = bias[col];
#pragma unroll
            for (int v = 0; v < 8; ++v) {
                const int row = bm + wm + mi * 16 + half * 8 + v;  // nb*1024+s
                const int nb  = row >> 10;
                const int s   = row & 1023;
                Out[((size_t)(s * 8 + nb)) * DMODEL + col] = acc[mi][ni][v] + bv;
            }
        }
}

// ---------------------------------------------------------------------------
// Flash attention: one wave per (stream, n, h, 16-query tile).
// Q/K/V are bf16, rows = s*N + n, head slice = h*64. Online softmax; P tile
// transposed C-layout -> A-layout through a per-wave LDS patch (s_wait_dscnt).
// ---------------------------------------------------------------------------
__global__ __launch_bounds__(256) void attn_kernel(const unsigned short* __restrict__ Qp,
                                                   const unsigned short* __restrict__ Qs,
                                                   const unsigned short* __restrict__ Kh,
                                                   const unsigned short* __restrict__ Vh,
                                                   const unsigned char* __restrict__ mask,
                                                   unsigned short* __restrict__ Op,
                                                   unsigned short* __restrict__ Os) {
    __shared__ unsigned short Plds[8][16 * LDA];   // 10 KB: per-wave 16x32 patch

    const int tid  = threadIdx.x;
    const int wave = tid >> 5;
    const int lane = tid & 31;
    const int half = lane >> 4;
    const int l16  = lane & 15;

    const int gw  = blockIdx.x * 8 + wave;    // 0..16383
    const int qt  = gw & 63;
    const int h   = (gw >> 6) & 15;
    const int n   = (gw >> 10) & 7;
    const int str = gw >> 13;

    const unsigned short* Q   = str ? Qs : Qp;
    unsigned short*       Out = str ? Os : Op;

    BFrag aq[2];   // Q A-fragments, invariant over key loop (DK=64 -> 2 chunks)
    {
        const int q = qt * 16 + l16;
        const unsigned short* qp = Q + ((size_t)q * 8 + n) * DMODEL + h * 64;
#pragma unroll
        for (int c = 0; c < 2; ++c) {
            aq[c].q[0] = *(const uint4*)(qp + c * 32 + half * 16);
            aq[c].q[1] = *(const uint4*)(qp + c * 32 + half * 16 + 8);
        }
    }

    float rowmax[8], rowsum[8];
#pragma unroll
    for (int v = 0; v < 8; ++v) { rowmax[v] = -3.0e38f; rowsum[v] = 0.0f; }
    v8f o[4] = {};
    unsigned short* myP = &Plds[wave][0];

    for (int kt = 0; kt < 1024; kt += 32) {
        v8f sc[2];
        float mval[2];
#pragma unroll
        for (int j = 0; j < 2; ++j) {
            const int key = kt + j * 16 + l16;
            const unsigned short* kp = Kh + ((size_t)key * 8 + n) * DMODEL + h * 64;
            BFrag bk;
            v8f c = {};
            bk.q[0] = *(const uint4*)(kp + half * 16);
            bk.q[1] = *(const uint4*)(kp + half * 16 + 8);
            c = wmma_bf16(aq[0], bk, c);
            bk.q[0] = *(const uint4*)(kp + 32 + half * 16);
            bk.q[1] = *(const uint4*)(kp + 32 + half * 16 + 8);
            c = wmma_bf16(aq[1], bk, c);
            sc[j] = c;
            mval[j] = mask[n * 1024 + key] ? -3.0e38f : 0.0f;
        }
#pragma unroll
        for (int j = 0; j < 2; ++j)
#pragma unroll
            for (int v = 0; v < 8; ++v)
                sc[j][v] = sc[j][v] * 0.125f + mval[j];   // 1/sqrt(64) + mask

        // online softmax; rows r = half*8+v live across a 16-lane half
#pragma unroll
        for (int v = 0; v < 8; ++v) {
            float tmax = fmaxf(sc[0][v], sc[1][v]);
#pragma unroll
            for (int m = 1; m < 16; m <<= 1)
                tmax = fmaxf(tmax, __shfl_xor(tmax, m, 32));
            const float nm    = fmaxf(rowmax[v], tmax);
            const float alpha = __expf(rowmax[v] - nm);
            rowmax[v] = nm;
            const float p0 = __expf(sc[0][v] - nm);
            const float p1 = __expf(sc[1][v] - nm);
            sc[0][v] = p0; sc[1][v] = p1;
            float ps = p0 + p1;
#pragma unroll
            for (int m = 1; m < 16; m <<= 1)
                ps += __shfl_xor(ps, m, 32);
            rowsum[v] = rowsum[v] * alpha + ps;
#pragma unroll
            for (int t = 0; t < 4; ++t) o[t][v] *= alpha;
        }

        // transpose P (C-layout -> A-layout) through per-wave LDS patch
#pragma unroll
        for (int j = 0; j < 2; ++j)
#pragma unroll
            for (int v = 0; v < 8; ++v)
                myP[(half * 8 + v) * LDA + j * 16 + l16] = f2bf(sc[j][v]);
        asm volatile("s_wait_dscnt 0" ::: "memory");

        BFrag ap;
        ap.q[0] = *(const uint4*)&myP[l16 * LDA + half * 16];
        ap.q[1] = *(const uint4*)&myP[l16 * LDA + half * 16 + 8];

        // O += P @ V : 4 dk-tiles of 16, contraction over 32 keys
#pragma unroll
        for (int t = 0; t < 4; ++t) {
            BFrag bv;
#pragma unroll
            for (int kk = 0; kk < 16; ++kk) {
                const int key = kt + half * 16 + kk;
                bv.u[kk] = Vh[((size_t)key * 8 + n) * DMODEL + h * 64 + t * 16 + l16];
            }
            o[t] = wmma_bf16(ap, bv, o[t]);
        }
    }

#pragma unroll
    for (int v = 0; v < 8; ++v) {
        const float inv = 1.0f / rowsum[v];
        const int s = qt * 16 + half * 8 + v;
        const size_t base = ((size_t)n * 1024 + s) * DMODEL + h * 64;
#pragma unroll
        for (int t = 0; t < 4; ++t)
            Out[base + t * 16 + l16] = f2bf(o[t][v] * inv);
    }
}

// ---------------------------------------------------------------------------
extern "C" void kernel_launch(void* const* d_in, const int* in_sizes, int n_in,
                              void* d_out, int out_size, void* d_ws, size_t ws_size,
                              hipStream_t stream) {
    const float*         Qpoi = (const float*)d_in[0];
    const float*         Qsvi = (const float*)d_in[1];
    const float*         Kin  = (const float*)d_in[2];
    const float*         Vin  = (const float*)d_in[3];
    const unsigned char* mask = (const unsigned char*)d_in[4];
    const float* wqp = (const float*)d_in[5];  const float* bqp = (const float*)d_in[6];
    const float* wqs = (const float*)d_in[7];  const float* bqs = (const float*)d_in[8];
    const float* wk  = (const float*)d_in[9];  const float* bk  = (const float*)d_in[10];
    const float* wv  = (const float*)d_in[11]; const float* bv  = (const float*)d_in[12];
    const float* wfc = (const float*)d_in[13]; const float* bfc = (const float*)d_in[14];
    float* out = (float*)d_out;

    const size_t T  = (size_t)8192 * 1024;         // activation tensor elements
    const size_t WT = (size_t)1024 * 1024;         // weight tensor elements
    unsigned short* ws = (unsigned short*)d_ws;    // bf16 workspace
    unsigned short* Qp = ws;                       // projected Q (poi)
    unsigned short* Qs = Qp + T;                   // projected Q (svi)
    unsigned short* Kw = Qs + T;                   // projected K
    unsigned short* Vw = Kw + T;                   // projected V
    unsigned short* Ap = Vw + T;                   // attention out (poi)
    unsigned short* As = Ap + T;                   // attention out (svi)
    unsigned short* Xb = As + T;                   // reusable bf16 activation buf
    unsigned short* Wb = Xb + T;                   // reusable bf16 weight buf

    dim3 gg(8, 64), bb(256);
    const int nAct4 = (int)(T / 4), nW4 = (int)(WT / 4);

    // Q_poi projection
    cvt_f32_bf16<<<2048, 256, 0, stream>>>(Qpoi, Xb, nAct4);
    cvt_f32_bf16<<<1024, 256, 0, stream>>>(wqp, Wb, nW4);
    gemm_proj<<<gg, bb, 0, stream>>>(Xb, Wb, bqp, Qp);
    // Q_svi projection
    cvt_f32_bf16<<<2048, 256, 0, stream>>>(Qsvi, Xb, nAct4);
    cvt_f32_bf16<<<1024, 256, 0, stream>>>(wqs, Wb, nW4);
    gemm_proj<<<gg, bb, 0, stream>>>(Xb, Wb, bqs, Qs);
    // K projection
    cvt_f32_bf16<<<2048, 256, 0, stream>>>(Kin, Xb, nAct4);
    cvt_f32_bf16<<<1024, 256, 0, stream>>>(wk, Wb, nW4);
    gemm_proj<<<gg, bb, 0, stream>>>(Xb, Wb, bk, Kw);
    // V projection
    cvt_f32_bf16<<<2048, 256, 0, stream>>>(Vin, Xb, nAct4);
    cvt_f32_bf16<<<1024, 256, 0, stream>>>(wv, Wb, nW4);
    gemm_proj<<<gg, bb, 0, stream>>>(Xb, Wb, bv, Vw);

    // dual attention (both streams in one launch)
    attn_kernel<<<2048, 256, 0, stream>>>(Qp, Qs, Kw, Vw, mask, Ap, As);

    // output projection + [N,S,D]->[S,N,D] transpose folded into stores
    cvt_f32_bf16<<<1024, 256, 0, stream>>>(wfc, Wb, nW4);
    gemm_fc<<<gg, bb, 0, stream>>>(Ap, Wb, bfc, out);
    gemm_fc<<<gg, bb, 0, stream>>>(As, Wb, bfc, out + T);
}
